// SoftHypergraphConv_83288005804663
// MI455X (gfx1250) — matrix operand order repaired and checked by reading, not compiled, not run on hardware
//
#include <hip/hip_runtime.h>

// ---------------------------------------------------------------------------
// SoftHypergraphConv on MI455X (gfx1250, wave32, WMMA, async global->LDS)
//
//   xtT  = (x @ W^T + b)^T                         (64 x 8192 bf16)
//   aggP = split-K partials of H^T @ x_t  (+d_e)   (S x 4096 x 64 f32)
//   aggT = (sum_s aggP * de_inv)^T                 (64 x 4096 bf16)
//   outP = split-K partials of H @ agg    (+d_n)   (S x 8192 x 64 f32)
//   out  = sum_s outP * dn_inv_sqrt                (8192 x 64 f32)
//
// Memory-bound problem (~8.6 GFLOP over 128 MB of H): H is streamed from HBM
// once (GEMM-2) and re-read from the 192 MB L2 (GEMM-3). Split-K (S=8) gives
// 512/1024 blocks so the chip's WGPs can actually saturate 23.3 TB/s.
// Degrees d_e/d_n are produced by a ones-matrix WMMA per K-step, landing
// directly in accumulator layout. Partials are private per block: no atomics,
// deterministic, no zero-init required.
// ---------------------------------------------------------------------------

typedef __attribute__((ext_vector_type(16))) __bf16 v16bf;
typedef __attribute__((ext_vector_type(8)))  float  v8f;

#define N_ROWS 8192
#define E_COLS 4096
#define C_DIM  64
#define LPITCH 40       // LDS row pitch in ushorts: 32 data + 8 pad = 80 B
#define KSPLIT 8        // split-K slices per GEMM

__device__ __forceinline__ unsigned short f2bf(float f) {
  return __builtin_bit_cast(unsigned short, (__bf16)f);   // RNE convert
}
__device__ __forceinline__ unsigned pk2(float a, float b) {
  return (unsigned)f2bf(a) | ((unsigned)f2bf(b) << 16);
}

// ---------------- kernel 1: xtT[c][n] = (x @ W^T + b)[n][c] ----------------
__global__ __launch_bounds__(256)
void xt_kernel(const float* __restrict__ x, const float* __restrict__ W,
               const float* __restrict__ b, unsigned short* __restrict__ xtT) {
  __shared__ float Ws[C_DIM][C_DIM];
  __shared__ float bs[C_DIM];
  const int tid = threadIdx.x;
  for (int i = tid; i < C_DIM * C_DIM; i += 256) Ws[i >> 6][i & 63] = W[i];
  if (tid < C_DIM) bs[tid] = b[tid];
  __syncthreads();
  const int n = blockIdx.x * 4 + (tid >> 6);
  const int c = tid & 63;
  const float* xr = x + (size_t)n * C_DIM;
  float acc = bs[c];
#pragma unroll
  for (int k = 0; k < C_DIM; ++k) acc += xr[k] * Ws[c][k];
  xtT[(size_t)c * N_ROWS + n] = f2bf(acc);
}

// ---------------- split-K GEMM + degree(WMMA), partial f32 output ----------
// MODE 0: P[s] += (H^T @ xt) slice   M = E_COLS, K = N_ROWS
// MODE 1: P[s] += (H   @ agg) slice  M = N_ROWS, K = E_COLS
template <int MODE>
__global__ __launch_bounds__(128)
void hg_gemm(const float* __restrict__ H, const unsigned short* __restrict__ BT,
             float* __restrict__ Pg, float* __restrict__ dPg) {
  constexpr int MT = 64;                       // output rows per block
  constexpr int KT = 32;                       // bf16 WMMA K depth
  constexpr int KTOT  = (MODE == 0) ? N_ROWS : E_COLS;
  constexpr int MROWS = (MODE == 0) ? E_COLS : N_ROWS;
  constexpr int KS    = KTOT / KSPLIT;         // K range per slice

  const int m0   = blockIdx.x * MT;
  const int s    = blockIdx.y;                 // split-K slice
  const int k0   = s * KS;
  const int tid  = threadIdx.x;
  const int lane = tid & 31;
  const int wv   = tid >> 5;                   // wave wv owns rows [16wv, 16wv+16)
  const int half = lane >> 4;
  const int l16  = lane & 15;

  __shared__ __align__(16) unsigned short Asm[MT][LPITCH];     // A  [m][k]
  __shared__ __align__(16) unsigned short Bsm[C_DIM][LPITCH];  // B^T[c][k]

  const unsigned bsm_base =
      (unsigned)(unsigned long long)(__attribute__((address_space(3))) unsigned short*)&Bsm[0][0];

  v8f acc[4] = {v8f{}, v8f{}, v8f{}, v8f{}};
  v8f accd   = {};                             // degree row-sums (WMMA with ones)

  union { v16bf v; unsigned short u[16]; } ones;
#pragma unroll
  for (int j = 0; j < 16; ++j) ones.u[j] = 0x3F80;  // bf16 1.0

  // async B copy: 2 threads per B^T row, 32-B chunk each, padded scatter in LDS
  const int bc_row = tid >> 1;                 // 0..63  (GEMM output column)
  const int bhalf  = tid & 1;                  // 0/1 -> 16-element chunk
  const unsigned laddrB = bsm_base + (unsigned)(bc_row * (LPITCH * 2) + bhalf * 32);

  for (int kb = k0; kb < k0 + KS; kb += KT) {
    __syncthreads();

    // ---- B tile: async global->LDS, per-lane addressed ----
    {
      unsigned long long gaddr = (unsigned long long)(const void*)
          (BT + (size_t)bc_row * KTOT + kb + bhalf * 16);
      asm volatile(
          "global_load_async_to_lds_b128 %0, %1, off\n"
          "global_load_async_to_lds_b128 %0, %1, off offset:16\n"
          :: "v"(laddrB), "v"(gaddr) : "memory");
    }

    // ---- A tile: f32 H -> bf16 into Asm[m][k] ----
    if (MODE == 0) {
      // A[m][k] = H[(kb+k)*E + m0+m]  (H^T): 4x4 register transpose
      const int kq = tid >> 4;                 // 0..7  -> k base kq*4
      const int mq = tid & 15;                 // 0..15 -> m base mq*4
      const float* src = H + (size_t)(kb + kq * 4) * E_COLS + (m0 + mq * 4);
      __builtin_prefetch(src + (size_t)KT * E_COLS, 0, 1);
      float4 r0 = ((const float4*)(src + 0 * E_COLS))[0];
      float4 r1 = ((const float4*)(src + 1 * E_COLS))[0];
      float4 r2 = ((const float4*)(src + 2 * E_COLS))[0];
      float4 r3 = ((const float4*)(src + 3 * E_COLS))[0];
      uint2 q;
      q.x = pk2(r0.x, r1.x); q.y = pk2(r2.x, r3.x);
      *(uint2*)&Asm[mq * 4 + 0][kq * 4] = q;
      q.x = pk2(r0.y, r1.y); q.y = pk2(r2.y, r3.y);
      *(uint2*)&Asm[mq * 4 + 1][kq * 4] = q;
      q.x = pk2(r0.z, r1.z); q.y = pk2(r2.z, r3.z);
      *(uint2*)&Asm[mq * 4 + 2][kq * 4] = q;
      q.x = pk2(r0.w, r1.w); q.y = pk2(r2.w, r3.w);
      *(uint2*)&Asm[mq * 4 + 3][kq * 4] = q;
    } else {
      // A[m][k] = H[(m0+m)*E + kb+k]: contiguous in k -> 2 x b128 LDS stores
      const int m = tid >> 1, h2 = tid & 1;
      const float* src = H + (size_t)(m0 + m) * E_COLS + (kb + h2 * 16);
      __builtin_prefetch(src + KT, 0, 1);
      float4 f0 = ((const float4*)src)[0];
      float4 f1 = ((const float4*)src)[1];
      float4 f2 = ((const float4*)src)[2];
      float4 f3 = ((const float4*)src)[3];
      uint4 p;
      p.x = pk2(f0.x, f0.y); p.y = pk2(f0.z, f0.w);
      p.z = pk2(f1.x, f1.y); p.w = pk2(f1.z, f1.w);
      *(uint4*)&Asm[m][h2 * 16 + 0] = p;
      p.x = pk2(f2.x, f2.y); p.y = pk2(f2.z, f2.w);
      p.z = pk2(f3.x, f3.y); p.w = pk2(f3.z, f3.w);
      *(uint4*)&Asm[m][h2 * 16 + 8] = p;
    }

    asm volatile("s_wait_asynccnt 0" ::: "memory");
    __syncthreads();

    // ---- A fragment: half 0 -> K{0..7,16..23}, half 1 -> K{8..15,24..31}
    union { v16bf v; uint4 q[2]; } af;
    const int am = wv * 16 + l16;
    af.q[0] = *(const uint4*)&Asm[am][8 * half];
    af.q[1] = *(const uint4*)&Asm[am][16 + 8 * half];

    // ---- all 4 B fragments first (one dscnt wait), then 5 WMMAs ----
    union { v16bf v; uint4 q[2]; } bfr[4];
#pragma unroll
    for (int t = 0; t < 4; ++t) {
      const int bc = t * 16 + l16;
      bfr[t].q[0] = *(const uint4*)&Bsm[bc][8 * half];
      bfr[t].q[1] = *(const uint4*)&Bsm[bc][16 + 8 * half];
    }

    accd = __builtin_amdgcn_wmma_f32_16x16x32_bf16(
        false, af.v, false, ones.v, (short)0, accd, false, false);
#pragma unroll
    for (int t = 0; t < 4; ++t)
      acc[t] = __builtin_amdgcn_wmma_f32_16x16x32_bf16(
          false, af.v, false, bfr[t].v, (short)0, acc[t], false, false);
  }

  // ---- epilogue: private partial tile + partial degree rows (no atomics) ----
  float* Pf = Pg + (size_t)s * MROWS * C_DIM;
#pragma unroll
  for (int i = 0; i < 8; ++i) {
    const int r = m0 + wv * 16 + 8 * half + i;
#pragma unroll
    for (int t = 0; t < 4; ++t)
      Pf[(size_t)r * C_DIM + t * 16 + l16] = acc[t][i];
  }
  if (l16 == 0) {                              // accd identical across columns
    float* dP = dPg + (size_t)s * MROWS;
#pragma unroll
    for (int i = 0; i < 8; ++i)
      dP[m0 + wv * 16 + 8 * half + i] = accd[i];
  }
}

// ---------------- finalize: reduce split-K partials + scale ----------------
__global__ __launch_bounds__(256)
void fin_agg(const float* __restrict__ P, const float* __restrict__ dP,
             unsigned short* __restrict__ aggT) {
  const int tid = threadIdx.x;
  const int e = blockIdx.x * 4 + (tid >> 6);
  const int c = tid & 63;
  float sum = 0.f, deg = 0.f;
#pragma unroll
  for (int s = 0; s < KSPLIT; ++s) {
    sum += P[(size_t)s * E_COLS * C_DIM + (size_t)e * C_DIM + c];
    deg += dP[(size_t)s * E_COLS + e];
  }
  const float sc = deg > 0.f ? __builtin_amdgcn_rcpf(deg) : 0.f;   // de_inv
  aggT[(size_t)c * E_COLS + e] = f2bf(sum * sc);
}

__global__ __launch_bounds__(256)
void fin_out(const float* __restrict__ P, const float* __restrict__ dP,
             float* __restrict__ out) {
  const int tid = threadIdx.x;
  const int n = blockIdx.x * 4 + (tid >> 6);
  const int c = tid & 63;
  float sum = 0.f, deg = 0.f;
#pragma unroll
  for (int s = 0; s < KSPLIT; ++s) {
    sum += P[(size_t)s * N_ROWS * C_DIM + (size_t)n * C_DIM + c];
    deg += dP[(size_t)s * N_ROWS + n];
  }
  const float sc = deg > 0.f ? __builtin_amdgcn_rsqf(deg) : 0.f;   // dn_inv_sqrt
  out[(size_t)n * C_DIM + c] = sum * sc;
}

// ---------------------------------------------------------------------------
extern "C" void kernel_launch(void* const* d_in, const int* in_sizes, int n_in,
                              void* d_out, int out_size, void* d_ws, size_t ws_size,
                              hipStream_t stream) {
  const float* x = (const float*)d_in[0];   // (8192, 64)
  const float* H = (const float*)d_in[1];   // (8192, 4096)
  const float* W = (const float*)d_in[2];   // (64, 64)
  const float* b = (const float*)d_in[3];   // (64,)

  char* ws = (char*)d_ws;
  size_t off = 0;
  unsigned short* xtT  = (unsigned short*)(ws + off); off += (size_t)C_DIM * N_ROWS * 2;  // 1 MB
  unsigned short* aggT = (unsigned short*)(ws + off); off += (size_t)C_DIM * E_COLS * 2;  // 0.5 MB
  float* aggP = (float*)(ws + off); off += (size_t)KSPLIT * E_COLS * C_DIM * 4;           // 8 MB
  float* deP  = (float*)(ws + off); off += (size_t)KSPLIT * E_COLS * 4;                   // 128 KB
  float* outP = (float*)(ws + off); off += (size_t)KSPLIT * N_ROWS * C_DIM * 4;           // 16 MB
  float* dnP  = (float*)(ws + off);                                                       // 256 KB

  xt_kernel<<<N_ROWS / 4, 256, 0, stream>>>(x, W, b, xtT);

  dim3 g2(E_COLS / 64, KSPLIT);
  hg_gemm<0><<<g2, 128, 0, stream>>>(H, xtT, aggP, deP);
  fin_agg<<<E_COLS / 4, 256, 0, stream>>>(aggP, deP, aggT);

  dim3 g3(N_ROWS / 64, KSPLIT);
  hg_gemm<1><<<g3, 128, 0, stream>>>(H, aggT, outP, dnP);
  fin_out<<<N_ROWS / 4, 256, 0, stream>>>(outP, dnP, (float*)d_out);
}
